// SelfAttention_55164559949933
// MI455X (gfx1250) — compile-verified, matrix operand order checked by reading, not run on hardware
//
#include <hip/hip_runtime.h>
#include <hip/hip_bf16.h>

// ---------------------------------------------------------------------------
// Self-attention (B=8, S=2048, d=1024) for gfx1250 (MI455X), wave32 WMMA.
//   1) Q = q@Wq^T+bq  -> bf16
//   2) K = k@Wk^T+bk  -> bf16
//   3) V = v@Wv^T+bv  -> bf16, stored TRANSPOSED per batch: Vt[b][n][s]
//   4) E = (Q@K^T)/32, query-row mask -> NEG   (f32, materialized; L2=192MB)
//   5) row stats: m = max, il = 1/sum(exp(e-m))
//   6) X = softmax(E)@V : exp applied in A-tile staging, *il in epilogue
//   7) out = X@Wfc^T+bfc -> f32 (d_out, nontemporal stores)
// GEMM core: 128x128x32 block, 8 waves (2x4), 4x2 wmma tiles/wave,
// double-buffered LDS; bf16-source tiles staged with
// GLOBAL_LOAD_ASYNC_TO_LDS_B128 (ASYNCcnt), one barrier pair per K-step.
// ---------------------------------------------------------------------------

typedef __attribute__((ext_vector_type(16))) __bf16 v16bf;
typedef __attribute__((ext_vector_type(8)))  float  v8f;

#define S_LEN   2048
#define D_DIM   1024
#define BATCH   8
#define BM      128
#define BN      128
#define BK      32
#define LDS_STR 40            // padded LDS row stride in bf16 elements (80B)
#define NEG_BIG (-1e10f)
#define INV_SCALE 0.03125f    // 1/sqrt(1024)

#if __has_builtin(__builtin_amdgcn_global_load_async_to_lds_b128)
#define HAVE_ASYNC_LDS 1
#else
#define HAVE_ASYNC_LDS 0
#endif

// GCC-style v4i matching the builtin prototype ("V4i*1", "V4i*3").
typedef int v4i __attribute__((vector_size(16)));
typedef __attribute__((address_space(1))) v4i v4i_gl;   // global
typedef __attribute__((address_space(3))) v4i v4i_sh;   // LDS

// ---- native bf16 converts (lower to v_cvt_*bf16_f32 on gfx1250) -----------
__device__ __forceinline__ unsigned short f2bf(float f) {
    __bf16 h = (__bf16)f;
    return __builtin_bit_cast(unsigned short, h);
}
__device__ __forceinline__ unsigned pack2(float lo, float hi) {
    return (unsigned)f2bf(lo) | ((unsigned)f2bf(hi) << 16);
}

// ---- async global->LDS 16B copy (falls back to manual copy) ----------------
__device__ __forceinline__ void async_copy16B(const __bf16* gsrc, __bf16* ldst) {
#if HAVE_ASYNC_LDS
    v4i* g = (v4i*)gsrc;          // generic reinterpret
    v4i* l = (v4i*)ldst;
    __builtin_amdgcn_global_load_async_to_lds_b128(
        (v4i_gl*)g, (v4i_sh*)l, 0, 0);
#else
    *(uint4*)ldst = *(const uint4*)gsrc;
#endif
}
__device__ __forceinline__ void wait_async_lds() {
#if HAVE_ASYNC_LDS
#if __has_builtin(__builtin_amdgcn_s_wait_asynccnt)
    __builtin_amdgcn_s_wait_asynccnt(0);
#else
    asm volatile("s_wait_asynccnt 0x0" ::: "memory");
#endif
#endif
}

union FragU { v16bf v; uint4 q[2]; };

// Fragment load matching the 16-bit A/B operand layout:
// lanes 0-15: rows 0-15, K = {0..7, 16..23}; lanes 16-31: K = {8..15, 24..31}
__device__ __forceinline__ v16bf load_frag(const __bf16* tile, int row0, int lane) {
    const int r  = row0 + (lane & 15);
    const int kh = (lane >> 4) << 3;          // 0 or 8
    FragU f;
    f.q[0] = *(const uint4*)(tile + r * LDS_STR + kh);
    f.q[1] = *(const uint4*)(tile + r * LDS_STR + 16 + kh);
    return f.v;
}

// --- 16-element (one 128x32-tile slice per thread) LDS stagers --------------
// Direct float4-component access: converts consume loaded registers in place
// (avoids the v_dual_mov shuffle storm seen with array aliasing).
__device__ __forceinline__ void stage_f32(const float* src, __bf16* dst) {
    const float4* p4 = (const float4*)src;
    const float4 f0 = p4[0], f1 = p4[1], f2 = p4[2], f3 = p4[3];
    *(uint4*)dst       = make_uint4(pack2(f0.x, f0.y), pack2(f0.z, f0.w),
                                    pack2(f1.x, f1.y), pack2(f1.z, f1.w));
    *(uint4*)(dst + 8) = make_uint4(pack2(f2.x, f2.y), pack2(f2.z, f2.w),
                                    pack2(f3.x, f3.y), pack2(f3.z, f3.w));
}
__device__ __forceinline__ void stage_exp(const float* src, float m, __bf16* dst) {
    const float4* p4 = (const float4*)src;
    float4 f0 = p4[0], f1 = p4[1], f2 = p4[2], f3 = p4[3];
    f0.x = __expf(f0.x - m); f0.y = __expf(f0.y - m);
    f0.z = __expf(f0.z - m); f0.w = __expf(f0.w - m);
    f1.x = __expf(f1.x - m); f1.y = __expf(f1.y - m);
    f1.z = __expf(f1.z - m); f1.w = __expf(f1.w - m);
    f2.x = __expf(f2.x - m); f2.y = __expf(f2.y - m);
    f2.z = __expf(f2.z - m); f2.w = __expf(f2.w - m);
    f3.x = __expf(f3.x - m); f3.y = __expf(f3.y - m);
    f3.z = __expf(f3.z - m); f3.w = __expf(f3.w - m);
    *(uint4*)dst       = make_uint4(pack2(f0.x, f0.y), pack2(f0.z, f0.w),
                                    pack2(f1.x, f1.y), pack2(f1.z, f1.w));
    *(uint4*)(dst + 8) = make_uint4(pack2(f2.x, f2.y), pack2(f2.z, f2.w),
                                    pack2(f3.x, f3.y), pack2(f3.z, f3.w));
}

// ---------------------------------------------------------------------------
// Templated WMMA GEMM: C[M,N] = A[M,K] * B^T[N,K] (+epilogue)
// ASRC: 0=f32, 1=bf16(async), 2=f32 with exp(e - m_row)
// BSRC: 0=f32 (weight rows = B^T rows), 1=bf16(async)
// EPI : 0=bias->bf16, 1=bias->bf16 transposed (Vt), 2=scale+mask->f32 (E),
//       3=*il ->bf16 (X), 4=bias->f32 nontemporal (final output)
// ---------------------------------------------------------------------------
template <int ASRC, int BSRC, int EPI>
__global__ __launch_bounds__(256) void wmma_gemm(
    const void* __restrict__ Ag, const void* __restrict__ Bg,
    const float* __restrict__ bias, void* __restrict__ Cg,
    const int* __restrict__ mask,
    const float* __restrict__ stat_m, const float* __restrict__ stat_il,
    int K, int lda, int ldb, int ldc,
    long long aBatch, long long bBatch, long long cBatch)
{
    __shared__ __align__(16) __bf16 As[2][BM * LDS_STR];
    __shared__ __align__(16) __bf16 Bs[2][BN * LDS_STR];

    const int tid  = threadIdx.x;
    const int lane = tid & 31;
    const int wave = tid >> 5;
    const int wm   = (wave & 1) * 64;   // wave row offset within block
    const int wn   = (wave >> 1) * 32;  // wave col offset within block

    const int row0 = blockIdx.y * BM;
    const int col0 = blockIdx.x * BN;
    const int z    = blockIdx.z;

    // Batch-offset typed base pointers (strides in elements of each type).
    const float*  Af = (const float*)Ag  + (size_t)z * aBatch;
    const __bf16* Ah = (const __bf16*)Ag + (size_t)z * aBatch;
    const float*  Bf = (const float*)Bg  + (size_t)z * bBatch;
    const __bf16* Bh = (const __bf16*)Bg + (size_t)z * bBatch;
    float*          Cf = (float*)Cg          + (size_t)z * cBatch;
    unsigned short* Cb = (unsigned short*)Cg + (size_t)z * cBatch;
    const float* sm = stat_m  + (size_t)z * S_LEN;   // ASRC==2
    const float* si = stat_il + (size_t)z * S_LEN;   // EPI==3

    // Tile staging: thread t owns 16 contiguous K-elements of row (t>>1),
    // segment ((t&1)*16), for both the A tile and the B^T tile.
    const int ldRow = tid >> 1;
    const int ldSeg = (tid & 1) << 4;

    float mRow = 0.f;
    if constexpr (ASRC == 2) mRow = sm[row0 + ldRow];

    auto stageA = [&](int buf, int k0) {
        __bf16* dst = &As[buf][ldRow * LDS_STR + ldSeg];
        if constexpr (ASRC == 0) {
            stage_f32(Af + (size_t)(row0 + ldRow) * lda + k0 + ldSeg, dst);
        } else if constexpr (ASRC == 2) {
            stage_exp(Af + (size_t)(row0 + ldRow) * lda + k0 + ldSeg, mRow, dst);
        } else {
            const __bf16* src = Ah + (size_t)(row0 + ldRow) * lda + k0 + ldSeg;
            async_copy16B(src, dst);
            async_copy16B(src + 8, dst + 8);
        }
    };
    auto stageB = [&](int buf, int k0) {
        __bf16* dst = &Bs[buf][ldRow * LDS_STR + ldSeg];
        if constexpr (BSRC == 0) {
            stage_f32(Bf + (size_t)(col0 + ldRow) * ldb + k0 + ldSeg, dst);
        } else {
            const __bf16* src = Bh + (size_t)(col0 + ldRow) * ldb + k0 + ldSeg;
            async_copy16B(src, dst);
            async_copy16B(src + 8, dst + 8);
        }
    };

    v8f acc[4][2];
#pragma unroll
    for (int mt = 0; mt < 4; ++mt)
#pragma unroll
        for (int nt = 0; nt < 2; ++nt) { v8f zv = {}; acc[mt][nt] = zv; }

    const int nIter = K / BK;
    stageA(0, 0);
    stageB(0, 0);
    wait_async_lds();
    __syncthreads();

    for (int it = 0; it < nIter; ++it) {
        const int buf = it & 1;
        if (it + 1 < nIter) {               // stage next tile into other buffer
            stageA(buf ^ 1, (it + 1) * BK);
            stageB(buf ^ 1, (it + 1) * BK);
        }
        v16bf a[4], b[2];
#pragma unroll
        for (int mt = 0; mt < 4; ++mt) a[mt] = load_frag(As[buf], wm + mt * 16, lane);
#pragma unroll
        for (int nt = 0; nt < 2; ++nt) b[nt] = load_frag(Bs[buf], wn + nt * 16, lane);
#pragma unroll
        for (int mt = 0; mt < 4; ++mt)
#pragma unroll
            for (int nt = 0; nt < 2; ++nt)
                acc[mt][nt] = __builtin_amdgcn_wmma_f32_16x16x32_bf16(
                    false, a[mt], false, b[nt], (short)0, acc[mt][nt], false, false);
        wait_async_lds();                   // next-tile LDS writes landed
        __syncthreads();                    // (compiler adds dscnt wait)
    }

    // ---- epilogue: lane l holds (M = vgpr + (l>=16 ? 8:0), N = l&15) ----
    const int colL = lane & 15;
    const int rowH = (lane >> 4) << 3;
#pragma unroll
    for (int mt = 0; mt < 4; ++mt) {
#pragma unroll
        for (int nt = 0; nt < 2; ++nt) {
            v8f c = acc[mt][nt];
            const int colG = col0 + wn + nt * 16 + colL;
            const int rowB = row0 + wm + mt * 16 + rowH;
#pragma unroll
            for (int i = 0; i < 8; ++i) {
                const int r = rowB + i;
                const float v = c[i];
                if constexpr (EPI == 0) {
                    Cb[(size_t)r * ldc + colG] = f2bf(v + bias[colG]);
                } else if constexpr (EPI == 1) {           // Vt[b][n][s]
                    const int bb = r >> 11;                // r / S_LEN
                    const int ss = r & (S_LEN - 1);
                    Cb[((size_t)bb * D_DIM + colG) * S_LEN + ss] = f2bf(v + bias[colG]);
                } else if constexpr (EPI == 2) {           // energy + mask
                    float e = v * INV_SCALE;
                    if (mask[(size_t)z * S_LEN + r] == 0) e = NEG_BIG;
                    Cf[(size_t)r * ldc + colG] = e;
                } else if constexpr (EPI == 3) {           // X = P@V * 1/l
                    Cb[(size_t)r * ldc + colG] = f2bf(v * si[r]);
                } else {                                   // final f32 + bias, NT
                    __builtin_nontemporal_store(v + bias[colG],
                                                &Cf[(size_t)r * ldc + colG]);
                }
            }
        }
    }
}

// ---------------------------------------------------------------------------
// Row softmax stats over E: m = rowmax, il = 1/sum(exp(e-m)).
// One 256-thread block per row (2048 elements, 8 per thread, kept in regs).
// Masked query rows (E == NEG everywhere) fall out naturally: il = 1/2048.
// ---------------------------------------------------------------------------
__global__ __launch_bounds__(256) void row_stats(const float* __restrict__ E,
                                                 float* __restrict__ m_out,
                                                 float* __restrict__ il_out)
{
    const int r   = blockIdx.x;
    const int tid = threadIdx.x;
    const float* row = E + (size_t)r * S_LEN;

    float vals[8];
    float mx = -3.0e38f;
#pragma unroll
    for (int i = 0; i < 8; ++i) {
        vals[i] = row[tid + i * 256];
        mx = fmaxf(mx, vals[i]);
    }
    __shared__ float red[256];
    red[tid] = mx;
    __syncthreads();
    for (int s = 128; s > 0; s >>= 1) {
        if (tid < s) red[tid] = fmaxf(red[tid], red[tid + s]);
        __syncthreads();
    }
    mx = red[0];
    __syncthreads();

    float sum = 0.f;
#pragma unroll
    for (int i = 0; i < 8; ++i) sum += __expf(vals[i] - mx);
    red[tid] = sum;
    __syncthreads();
    for (int s = 128; s > 0; s >>= 1) {
        if (tid < s) red[tid] += red[tid + s];
        __syncthreads();
    }
    if (tid == 0) {
        m_out[r]  = mx;
        il_out[r] = 1.0f / red[0];
    }
}

// ---------------------------------------------------------------------------
extern "C" void kernel_launch(void* const* d_in, const int* in_sizes, int n_in,
                              void* d_out, int out_size, void* d_ws, size_t ws_size,
                              hipStream_t stream)
{
    const float* query = (const float*)d_in[0];
    const float* key_  = (const float*)d_in[1];
    const float* value = (const float*)d_in[2];
    const int*   mask  = (const int*)d_in[3];
    const float* Wq  = (const float*)d_in[4];   const float* bq  = (const float*)d_in[5];
    const float* Wk  = (const float*)d_in[6];   const float* bk  = (const float*)d_in[7];
    const float* Wv  = (const float*)d_in[8];   const float* bv  = (const float*)d_in[9];
    const float* Wfc = (const float*)d_in[10];  const float* bfc = (const float*)d_in[11];

    const size_t BS = (size_t)BATCH * S_LEN;            // 16384 rows
    char* ws = (char*)d_ws;
    // Workspace layout (~225MB):
    //   [0,32MB)    Qbf (bf16)  -- reused as X after energy GEMM
    //   [32,64MB)   Kbf (bf16)
    //   [64,96MB)   Vt  (bf16, per-batch transposed V: [b][n][s])
    //   [96,224MB)  E   (f32, B*S*S)
    //   [224MB,..)  stats: m[BS], il[BS]
    __bf16* Qbf = (__bf16*)ws;
    __bf16* Kbf = (__bf16*)(ws + BS * D_DIM * 2);
    __bf16* Vt  = (__bf16*)(ws + BS * D_DIM * 4);
    float*  E   = (float*) (ws + BS * D_DIM * 6);
    float*  stat_m  = (float*)(ws + BS * D_DIM * 6 + (size_t)BATCH * S_LEN * S_LEN * 4);
    float*  stat_il = stat_m + BS;
    __bf16* X = Qbf;   // reuse: Q dead after energy GEMM

    const dim3 blk(256);
    const long long sd = (long long)S_LEN * D_DIM;
    const long long ss = (long long)S_LEN * S_LEN;

    // 1-3) projections
    wmma_gemm<0, 0, 0><<<dim3(D_DIM / BN, BS / BM, 1), blk, 0, stream>>>(
        query, Wq, bq, Qbf, nullptr, stat_m, stat_il,
        D_DIM, D_DIM, D_DIM, D_DIM, 0, 0, 0);
    wmma_gemm<0, 0, 0><<<dim3(D_DIM / BN, BS / BM, 1), blk, 0, stream>>>(
        key_, Wk, bk, Kbf, nullptr, stat_m, stat_il,
        D_DIM, D_DIM, D_DIM, D_DIM, 0, 0, 0);
    wmma_gemm<0, 0, 1><<<dim3(D_DIM / BN, BS / BM, 1), blk, 0, stream>>>(
        value, Wv, bv, Vt, nullptr, stat_m, stat_il,
        D_DIM, D_DIM, D_DIM, D_DIM, 0, 0, 0);

    // 4) energy E[b] = Q[b] @ K[b]^T / 32, query-row mask
    wmma_gemm<1, 1, 2><<<dim3(S_LEN / BN, S_LEN / BM, BATCH), blk, 0, stream>>>(
        Qbf, Kbf, nullptr, E, mask, stat_m, stat_il,
        D_DIM, D_DIM, D_DIM, S_LEN, sd, sd, ss);

    // 5) row softmax stats
    row_stats<<<dim3((unsigned)BS), blk, 0, stream>>>(E, stat_m, stat_il);

    // 6) X[b] = softmax(E[b]) @ V[b]
    wmma_gemm<2, 1, 3><<<dim3(D_DIM / BN, S_LEN / BM, BATCH), blk, 0, stream>>>(
        E, Vt, nullptr, X, nullptr, stat_m, stat_il,
        S_LEN, S_LEN, S_LEN, D_DIM, ss, sd, sd);

    // 7) out = X @ Wfc^T + bfc  (f32, nontemporal)
    wmma_gemm<1, 0, 4><<<dim3(D_DIM / BN, BS / BM, 1), blk, 0, stream>>>(
        X, Wfc, bfc, (float*)d_out, nullptr, stat_m, stat_il,
        D_DIM, D_DIM, D_DIM, D_DIM, 0, 0, 0);
}
